// GRU_20753281974856
// MI455X (gfx1250) — compile-verified, matrix operand order checked by reading, not run on hardware
//
#include <hip/hip_runtime.h>
#include <math.h>

// GRU on MI455X (gfx1250): FP16 WMMA (f32 accumulate) for all GEMMs.
// T=512, B=64, I=128, H=512 (fixed by the reference).
// All WMMA operands pre-staged in fp16 so the recurrent inner loops are
// pure global_load_b128 + v_wmma_f32_16x16x32_f16.

typedef __attribute__((ext_vector_type(16))) _Float16 v16h;
typedef __attribute__((ext_vector_type(8)))  _Float16 v8h;
typedef __attribute__((ext_vector_type(8)))  float    v8f;

#define T_STEPS 512
#define B_DIM   64
#define I_DIM   128
#define H_DIM   512

// ---------------------------------------------------------------------------
// WMMA fragment loaders (CDNA5 ISA layouts, wave32).
//
// A (16-bit, 16xK tile of 32 K-values starting at kb):
//   lane L: row = m0 + (L%16), half = L/16
//   elems 0..7  = A[row][kb + 8*half + j]
//   elems 8..15 = A[row][kb + 16 + 8*half + j]
// B (16-bit, 32x16): mirrored on columns; W stored transposed as [N][K]
//   so the same addressing gives two contiguous 16-byte runs.
// C/D (f32 16x16): elem v -> row m0 + v + 8*half, col n0 + (L%16).
// All fragment bases are 16B-aligned: ld ∈ {128,512}, kb ≡ 0 (mod 32),
// half-offset is 0 or 16 bytes.
// ---------------------------------------------------------------------------

__device__ __forceinline__ v16h frag_h16(const _Float16* __restrict__ M,
                                         int ld, int rc0, int kb, int lane) {
    const int rc   = rc0 + (lane & 15);
    const int half = (lane >> 4) & 1;
    const _Float16* p = M + (size_t)rc * ld + kb + 8 * half;
    const v8h lo = *(const v8h*)(p);
    const v8h hi = *(const v8h*)(p + 16);
    v16h r;
#pragma unroll
    for (int j = 0; j < 8; ++j) { r[j] = lo[j]; r[8 + j] = hi[j]; }
    return r;
}

__device__ __forceinline__ float fast_sigmoid(float x) {
    return 1.0f / (1.0f + __expf(-x));
}

// ---------------------------------------------------------------------------
// One-time conversions.
// ---------------------------------------------------------------------------

// src: [K][N] fp32 row-major -> dst: [N][K] fp16 row-major (transpose+cvt)
__global__ void gru_cvt_transpose(const float* __restrict__ src,
                                  _Float16* __restrict__ dst, int K, int N) {
    int idx = blockIdx.x * blockDim.x + threadIdx.x;
    if (idx >= K * N) return;
    int n = idx / K;
    int k = idx - n * K;
    dst[idx] = (_Float16)src[(size_t)k * N + n];
}

// straight fp32 -> fp16 convert (for X)
__global__ void gru_cvt_h16(const float* __restrict__ src,
                            _Float16* __restrict__ dst, int n) {
    int i = blockIdx.x * blockDim.x + threadIdx.x;
    if (i < n) dst[i] = (_Float16)src[i];
}

// ---------------------------------------------------------------------------
// Phase A (per step t): Z = sigmoid(X_t*Wxz + H*Whz + bz)
//                       R = sigmoid(X_t*Wxr + H*Whr + br); RH = R .* H
// 256 tiles (gate z: 128, gate r: 128); 1 tile per wave; 32 blocks x 8 waves.
// ---------------------------------------------------------------------------
__global__ void __launch_bounds__(256)
gru_phaseA(const _Float16* __restrict__ Xh16, const _Float16* __restrict__ Hh16,
           const _Float16* __restrict__ Whz_t, const _Float16* __restrict__ Whr_t,
           const _Float16* __restrict__ Wxz_t, const _Float16* __restrict__ Wxr_t,
           const float* __restrict__ b_z, const float* __restrict__ b_r,
           float* __restrict__ Zbuf, _Float16* __restrict__ RHbuf, int t) {
    const int lane = threadIdx.x & 31;
    const int wave = threadIdx.x >> 5;
    const int tid  = blockIdx.x * 8 + wave;   // 0..255
    const int gate = tid >> 7;                // 0 = z, 1 = r
    const int tile = tid & 127;
    const int m0   = (tile >> 5) * 16;        // batch tile (4 tiles)
    const int n0   = (tile & 31) * 16;        // hidden tile (32 tiles)

    const _Float16* Wh = gate ? Whr_t : Whz_t;
    const _Float16* Wx = gate ? Wxr_t : Wxz_t;
    const float*  bias = gate ? b_r  : b_z;

    v8f acc = {};
    if (t > 0) {
        for (int kb = 0; kb < H_DIM; kb += 32) {
            v16h a  = frag_h16(Hh16, H_DIM, m0, kb, lane);
            v16h bb = frag_h16(Wh,   H_DIM, n0, kb, lane);
            acc = __builtin_amdgcn_wmma_f32_16x16x32_f16(
                      false, a, false, bb, (short)0, acc, false, false);
        }
    }
    const _Float16* Xt = Xh16 + (size_t)t * B_DIM * I_DIM;
#pragma unroll
    for (int kb = 0; kb < I_DIM; kb += 32) {
        v16h a  = frag_h16(Xt, I_DIM, m0, kb, lane);
        v16h bb = frag_h16(Wx, I_DIM, n0, kb, lane);
        acc = __builtin_amdgcn_wmma_f32_16x16x32_f16(
                  false, a, false, bb, (short)0, acc, false, false);
    }

    const int col  = n0 + (lane & 15);
    const int half = (lane >> 4) & 1;
    const float bv = bias[col];
#pragma unroll
    for (int v = 0; v < 8; ++v) {
        const int row = m0 + v + 8 * half;
        const size_t idx = (size_t)row * H_DIM + col;
        const float g = fast_sigmoid(acc[v] + bv);
        if (gate == 0) {
            Zbuf[idx] = g;
        } else {
            const float h = (t > 0) ? (float)Hh16[idx] : 0.0f;
            RHbuf[idx] = (_Float16)(g * h);
        }
    }
}

// ---------------------------------------------------------------------------
// Phase B (per step t): Ht = tanh(X_t*Wxh + RH*Whh + bh)
//                       Hnew = Z.*H + (1-Z).*Ht  -> out[t] (fp32) + Hh16 (fp16)
// 128 tiles; 16 blocks x 8 waves.
// ---------------------------------------------------------------------------
__global__ void __launch_bounds__(256)
gru_phaseB(const _Float16* __restrict__ Xh16, float* __restrict__ out,
           _Float16* __restrict__ Hh16,
           const _Float16* __restrict__ Whh_t, const _Float16* __restrict__ Wxh_t,
           const float* __restrict__ b_h,
           const float* __restrict__ Zbuf, const _Float16* __restrict__ RHbuf,
           int t) {
    const int lane = threadIdx.x & 31;
    const int wave = threadIdx.x >> 5;
    const int tid  = blockIdx.x * 8 + wave;   // 0..127
    const int m0   = (tid >> 5) * 16;
    const int n0   = (tid & 31) * 16;

    v8f acc = {};
    for (int kb = 0; kb < H_DIM; kb += 32) {
        v16h a  = frag_h16(RHbuf, H_DIM, m0, kb, lane);
        v16h bb = frag_h16(Whh_t, H_DIM, n0, kb, lane);
        acc = __builtin_amdgcn_wmma_f32_16x16x32_f16(
                  false, a, false, bb, (short)0, acc, false, false);
    }
    const _Float16* Xt = Xh16 + (size_t)t * B_DIM * I_DIM;
#pragma unroll
    for (int kb = 0; kb < I_DIM; kb += 32) {
        v16h a  = frag_h16(Xt, I_DIM, m0, kb, lane);
        v16h bb = frag_h16(Wxh_t, I_DIM, n0, kb, lane);
        acc = __builtin_amdgcn_wmma_f32_16x16x32_f16(
                  false, a, false, bb, (short)0, acc, false, false);
    }

    const float* Hprev = (t > 0) ? (out + (size_t)(t - 1) * B_DIM * H_DIM) : nullptr;
    float* Hout = out + (size_t)t * B_DIM * H_DIM;
    const int col  = n0 + (lane & 15);
    const int half = (lane >> 4) & 1;
    const float bv = b_h[col];
#pragma unroll
    for (int v = 0; v < 8; ++v) {
        const int row = m0 + v + 8 * half;
        const size_t idx = (size_t)row * H_DIM + col;
        const float ht = tanhf(acc[v] + bv);
        const float z  = Zbuf[idx];
        const float hp = Hprev ? Hprev[idx] : 0.0f;
        const float hn = z * hp + (1.0f - z) * ht;
        Hout[idx] = hn;                 // fp32 recurrence state (accuracy path)
        Hh16[idx] = (_Float16)hn;       // fp16 WMMA operand for step t+1
    }
}

__global__ void gru_copy_final(const float* __restrict__ src,
                               float* __restrict__ dst, int n) {
    int i = blockIdx.x * blockDim.x + threadIdx.x;
    if (i < n) dst[i] = src[i];
}

// ---------------------------------------------------------------------------
extern "C" void kernel_launch(void* const* d_in, const int* in_sizes, int n_in,
                              void* d_out, int out_size, void* d_ws, size_t ws_size,
                              hipStream_t stream) {
    const float* X   = (const float*)d_in[0];
    const float* Wxz = (const float*)d_in[1];
    const float* Whz = (const float*)d_in[2];
    const float* bz  = (const float*)d_in[3];
    const float* Wxr = (const float*)d_in[4];
    const float* Whr = (const float*)d_in[5];
    const float* br  = (const float*)d_in[6];
    const float* Wxh = (const float*)d_in[7];
    const float* Whh = (const float*)d_in[8];
    const float* bh  = (const float*)d_in[9];
    float* out = (float*)d_out;   // [T][B][H] outputs, then [B][H] H_final

    // Workspace layout: fp32 Zbuf first (alignment), then fp16 arrays.
    float*    Zbuf  = (float*)d_ws;                               // B*H fp32
    _Float16* Whz_t = (_Float16*)(Zbuf + (size_t)B_DIM * H_DIM);  // H*H
    _Float16* Whr_t = Whz_t + (size_t)H_DIM * H_DIM;              // H*H
    _Float16* Whh_t = Whr_t + (size_t)H_DIM * H_DIM;              // H*H
    _Float16* Wxz_t = Whh_t + (size_t)H_DIM * H_DIM;              // H*I
    _Float16* Wxr_t = Wxz_t + (size_t)H_DIM * I_DIM;              // H*I
    _Float16* Wxh_t = Wxr_t + (size_t)H_DIM * I_DIM;              // H*I
    _Float16* RHbuf = Wxh_t + (size_t)H_DIM * I_DIM;              // B*H
    _Float16* Hh16  = RHbuf + (size_t)B_DIM * H_DIM;              // B*H
    _Float16* Xh16  = Hh16  + (size_t)B_DIM * H_DIM;              // T*B*I

    // One-time weight transpose/convert + X convert (L2-resident thereafter).
    {
        const int nHH = H_DIM * H_DIM, nIH = I_DIM * H_DIM, blk = 256;
        gru_cvt_transpose<<<(nHH + blk - 1) / blk, blk, 0, stream>>>(Whz, Whz_t, H_DIM, H_DIM);
        gru_cvt_transpose<<<(nHH + blk - 1) / blk, blk, 0, stream>>>(Whr, Whr_t, H_DIM, H_DIM);
        gru_cvt_transpose<<<(nHH + blk - 1) / blk, blk, 0, stream>>>(Whh, Whh_t, H_DIM, H_DIM);
        gru_cvt_transpose<<<(nIH + blk - 1) / blk, blk, 0, stream>>>(Wxz, Wxz_t, I_DIM, H_DIM);
        gru_cvt_transpose<<<(nIH + blk - 1) / blk, blk, 0, stream>>>(Wxr, Wxr_t, I_DIM, H_DIM);
        gru_cvt_transpose<<<(nIH + blk - 1) / blk, blk, 0, stream>>>(Wxh, Wxh_t, I_DIM, H_DIM);
        const int nX = T_STEPS * B_DIM * I_DIM;
        gru_cvt_h16<<<(nX + blk - 1) / blk, blk, 0, stream>>>(X, Xh16, nX);
    }

    // Sequential recurrence: two stream-ordered WMMA kernels per step.
    for (int t = 0; t < T_STEPS; ++t) {
        gru_phaseA<<<32, 256, 0, stream>>>(Xh16, Hh16, Whz_t, Whr_t, Wxz_t, Wxr_t,
                                           bz, br, Zbuf, RHbuf, t);
        gru_phaseB<<<16, 256, 0, stream>>>(Xh16, out, Hh16, Whh_t, Wxh_t, bh,
                                           Zbuf, RHbuf, t);
    }

    // H_final = out[T-1] appended after the [T,B,H] block.
    const int nBH = B_DIM * H_DIM;
    gru_copy_final<<<(nBH + 255) / 256, 256, 0, stream>>>(
        out + (size_t)(T_STEPS - 1) * nBH, out + (size_t)T_STEPS * nBH, nBH);
}
